// NequIPConvolution_82240033784027
// MI455X (gfx1250) — matrix operand order repaired and checked by reading, not compiled, not run on hardware
//
#include <hip/hip_runtime.h>
#include <hip/hip_bf16.h>
#include <math.h>

// ---------------------------------------------------------------------------
// Problem constants (fixed by the reference)
// ---------------------------------------------------------------------------
#define NN   10000
#define EE   320000
#define MUL  64
#define NSP  4
#define NB   8
#define NH   64
#define INV_S3 0.5773502691896258f

typedef __bf16 bf16_t;
typedef bf16_t v16bf __attribute__((ext_vector_type(16)));
typedef float  v8f   __attribute__((ext_vector_type(8)));

// silu via v_exp_f32 + v_rcp_f32 (TRANS ops, co-execute with WMMA) instead of
// the IEEE v_div_scale/div_fmas sequence the plain division lowers to.
__device__ __forceinline__ float silu_f(float x) {
    return x * __builtin_amdgcn_rcpf(1.0f + __expf(-x));
}

__device__ __forceinline__ v8f v8f_zero() {
    v8f c;
#pragma unroll
    for (int i = 0; i < 8; ++i) c[i] = 0.0f;
    return c;
}

__device__ __forceinline__ v8f wmma_bf16(v16bf a, v16bf b, v8f c) {
    // D(16x16,f32) = A(16x32,bf16) * B(32x16,bf16) + C
    return __builtin_amdgcn_wmma_f32_16x16x32_bf16(false, a, false, b, (short)0, c, false, false);
}

// ---------------------------------------------------------------------------
// Weight conversion: f32 row-major W[K][N] -> bf16 B-fragments.
// Fragment f = nt*Ktiles + kt holds the 32x16 tile (kt,nt).
// B layout (wave32, 16-bit):  lane L -> N = L&15,  K = kt*32 + ((j&8)<<1) + 8*(L>>4) + (j&7)
// Each lane's 16 elements are stored contiguously -> 32B vector loads later.
// ---------------------------------------------------------------------------
__global__ void convert_weights_kernel(const float* __restrict__ W, int Kreal,
                                       int Ktiles, int Ntiles, bf16_t* __restrict__ dst) {
    int total = Ktiles * Ntiles * 512;
    int Ncols = Ntiles * 16;
    for (int id = blockIdx.x * blockDim.x + threadIdx.x; id < total;
         id += gridDim.x * blockDim.x) {
        int f    = id >> 9;
        int rem  = id & 511;
        int lane = rem >> 4;
        int j    = rem & 15;
        int kt   = f % Ktiles;
        int nt   = f / Ktiles;
        int K    = kt * 32 + ((j & 8) << 1) + ((lane >> 4) << 3) + (j & 7);
        int Ncol = nt * 16 + (lane & 15);
        float v  = (K < Kreal) ? W[(size_t)K * Ncols + Ncol] : 0.0f;
        dst[id]  = (bf16_t)v;
    }
}

// ---------------------------------------------------------------------------
// Node pre-pass: h0 = x0@lin1_ws/8, h1 = x1(u,i)@lin1_wv/8,
//                sc_s/sc_v = species-selected self-connection / 16.
// ---------------------------------------------------------------------------
__global__ void __launch_bounds__(256) node_pre_kernel(
    const float* __restrict__ nf, const float* __restrict__ attr,
    const float* __restrict__ l1ws, const float* __restrict__ l1wv,
    const float* __restrict__ scws, const float* __restrict__ scwv,
    float* __restrict__ h0, float* __restrict__ h1,
    float* __restrict__ sc_s, float* __restrict__ sc_v) {
    const int n = blockIdx.x;
    __shared__ float sx[256];   // [0..63] = x0, [64+u*3+i] = x1
    __shared__ int ssp;
    const int t = threadIdx.x;
    sx[t] = nf[(size_t)n * 256 + t];
    if (t < 4) { if (attr[(size_t)n * 4 + t] > 0.5f) ssp = t; }
    __syncthreads();
    const int sp = ssp;
    const float inv = 0.125f;    // 1/sqrt(64)
    const float fan = 0.0625f;   // 1/sqrt(64*4)
    for (int o = t; o < 576; o += 256) {
        if (o < 64) {
            int v = o; float acc = 0.0f;
            for (int u = 0; u < 64; ++u) acc += sx[u] * l1ws[u * 64 + v];
            h0[(size_t)n * 64 + v] = acc * inv;
        } else if (o < 256) {
            int q = o - 64, v = q / 3, i = q % 3; float acc = 0.0f;
            for (int u = 0; u < 64; ++u) acc += sx[64 + u * 3 + i] * l1wv[u * 64 + v];
            h1[(size_t)n * 192 + v * 3 + i] = acc * inv;
        } else if (o < 384) {
            int w = o - 256; float acc = 0.0f;
            for (int u = 0; u < 64; ++u) acc += sx[u] * scws[u * 512 + sp * 128 + w];
            sc_s[(size_t)n * 128 + w] = acc * fan;
        } else {
            int q = o - 384, v = q / 3, i = q % 3; float acc = 0.0f;
            for (int u = 0; u < 64; ++u) acc += sx[64 + u * 3 + i] * scwv[u * 256 + sp * 64 + v];
            sc_v[(size_t)n * 192 + v * 3 + i] = acc * fan;
        }
    }
}

// ---------------------------------------------------------------------------
// Edge kernel: one wave owns a 16-edge tile.
//   MLP (bf16 WMMA, f32 accum): 4 + 8 + 32 = 44 v_wmma per wave.
//   A-fragment layout (16-bit A 16x32): lane L -> M = L&15,
//     element j -> K = ((j&8)<<1) + 8*(L>>4) + (j&7)   [= j + (j&8) + 8*half]
//   Activations staged in LDS rows of stride 72 bf16 (conflict-free, 16B rows).
//   Layer-3 consumed in two passes of paired 64-channel chunks so each gathered
//   node value (e0 for s00/v01, e1 for v10/s11) is loaded exactly once from the
//   L2-resident h0/h1; scatter f32 atomics into agg_s/agg_v (1/32 folded in).
// ---------------------------------------------------------------------------
__global__ void __launch_bounds__(256) edge_kernel(
    const float* __restrict__ edge_sh, const float* __restrict__ edge_emb,
    const int* __restrict__ edge_src, const int* __restrict__ edge_dst,
    const float* __restrict__ h0, const float* __restrict__ h1,
    const bf16_t* __restrict__ wf1, const bf16_t* __restrict__ wf2,
    const bf16_t* __restrict__ wf3,
    float* __restrict__ agg_s, float* __restrict__ agg_v) {
    __shared__ bf16_t s_act[8][16][72];
    __shared__ float  s_shv[8][16][4];
    __shared__ int    s_src[8][16];
    __shared__ int    s_dst[8][16];

    const int wv    = threadIdx.x >> 5;
    const int lane  = threadIdx.x & 31;
    const int m     = lane & 15;
    const int half  = lane >> 4;
    const int ebase = (blockIdx.x * 8 + wv) * 16;

    // ---- stage per-edge metadata (lanes 0..15) ----
    if (half == 0) {
        int e = ebase + m;
        s_src[wv][m] = edge_src[e];
        s_dst[wv][m] = edge_dst[e];
        const float4 sh4 = *(const float4*)(edge_sh + (size_t)e * 4);
        s_shv[wv][m][0] = sh4.x; s_shv[wv][m][1] = sh4.y;
        s_shv[wv][m][2] = sh4.z; s_shv[wv][m][3] = sh4.w;
    }

    // ---- layer 1: A = edge_embedded (16x8, zero-padded to 16x32) ----
    v16bf a0;
#pragma unroll
    for (int j = 0; j < 16; ++j) a0[j] = (bf16_t)0.0f;
    if (half == 0) {
        const float* ep = edge_emb + (size_t)(ebase + m) * 8;
#pragma unroll
        for (int j = 0; j < 8; ++j) a0[j] = (bf16_t)ep[j];   // K=j for j<8, half=0
    }
#pragma unroll
    for (int nt = 0; nt < 4; ++nt) {
        v16bf b = *(const v16bf*)(wf1 + (size_t)nt * 512 + lane * 16);
        v8f c = v8f_zero();
        c = wmma_bf16(a0, b, c);
#pragma unroll
        for (int r = 0; r < 8; ++r)
            s_act[wv][r + 8 * half][nt * 16 + m] = (bf16_t)silu_f(c[r]);
    }

    // ---- layer 2 ----
    v16bf a2[2];
#pragma unroll
    for (int kt = 0; kt < 2; ++kt)
#pragma unroll
        for (int j = 0; j < 16; ++j) {
            int kl = ((j & 8) << 1) + (half << 3) + (j & 7);
            a2[kt][j] = s_act[wv][m][kt * 32 + kl];
        }
#pragma unroll
    for (int nt = 0; nt < 4; ++nt) {
        v8f c = v8f_zero();
#pragma unroll
        for (int kt = 0; kt < 2; ++kt) {
            v16bf b = *(const v16bf*)(wf2 + (size_t)(nt * 2 + kt) * 512 + lane * 16);
            c = wmma_bf16(a2[kt], b, c);
        }
#pragma unroll
        for (int r = 0; r < 8; ++r)
            s_act[wv][r + 8 * half][nt * 16 + m] = (bf16_t)silu_f(c[r]);
    }

    // ---- layer 3 A-fragments ----
    v16bf a3[2];
#pragma unroll
    for (int kt = 0; kt < 2; ++kt)
#pragma unroll
        for (int j = 0; j < 16; ++j) {
            int kl = ((j & 8) << 1) + (half << 3) + (j & 7);
            a3[kt][j] = s_act[wv][m][kt * 32 + kl];
        }

    const float inv_nb = 1.0f / 32.0f;

    // ---- layer 3, pass p=0: chunks (w00,w01); pass p=1: chunks (w10,w11) ----
#pragma unroll
    for (int p = 0; p < 2; ++p) {
        v8f ccA[4], ccB[4];   // chunk 2p and chunk 2p+1
#pragma unroll
        for (int nt = 0; nt < 4; ++nt) {
            ccA[nt] = v8f_zero();
            ccB[nt] = v8f_zero();
#pragma unroll
            for (int kt = 0; kt < 2; ++kt) {
                v16bf bA = *(const v16bf*)(wf3 + (size_t)(((2 * p) * 4 + nt) * 2 + kt) * 512 + lane * 16);
                ccA[nt] = wmma_bf16(a3[kt], bA, ccA[nt]);
                v16bf bB = *(const v16bf*)(wf3 + (size_t)(((2 * p + 1) * 4 + nt) * 2 + kt) * 512 + lane * 16);
                ccB[nt] = wmma_bf16(a3[kt], bB, ccB[nt]);
            }
        }
        // merged epilogue: lane holds channel u = nt*16 + m for edges M = r+8*half
#pragma unroll
        for (int nt = 0; nt < 4; ++nt) {
            const int u = nt * 16 + m;
#pragma unroll
            for (int r = 0; r < 8; ++r) {
                const int   me   = r + 8 * half;
                const float wA   = ccA[nt][r];
                const float wB   = ccB[nt][r];
                const int   srcn = s_src[wv][me];
                const int   dstn = s_dst[wv][me];
                const float sh0v = s_shv[wv][me][0];
                if (p == 0) {
                    // s00 = w00*e0*sh0 ; v01 = (w01*e0) x sh1   (e0 loaded once)
                    float e0 = h0[(size_t)srcn * 64 + u];
                    atomicAdd(&agg_s[(size_t)dstn * 128 + u], wA * e0 * sh0v * inv_nb);
                    float tv = wB * e0 * inv_nb;
#pragma unroll
                    for (int i = 0; i < 3; ++i)
                        atomicAdd(&agg_v[(size_t)dstn * 384 + u * 3 + i],
                                  tv * s_shv[wv][me][1 + i]);
                } else {
                    // v10 = w10*e1*sh0 ; s11 = w11*(e1.sh1)/sqrt3  (e1 loaded once)
                    float tv  = wA * sh0v * inv_nb;
                    float dot = 0.0f;
#pragma unroll
                    for (int i = 0; i < 3; ++i) {
                        float e1v = h1[(size_t)srcn * 192 + u * 3 + i];
                        atomicAdd(&agg_v[(size_t)dstn * 384 + (64 + u) * 3 + i], tv * e1v);
                        dot += e1v * s_shv[wv][me][1 + i];
                    }
                    atomicAdd(&agg_s[(size_t)dstn * 128 + 64 + u],
                              wB * dot * INV_S3 * inv_nb);
                }
            }
        }
    }
}

// ---------------------------------------------------------------------------
// Node post-pass: out_s = agg_s@lin2_ws/sqrt(128) + sc_s ; gating ; out_v.
// ---------------------------------------------------------------------------
__global__ void __launch_bounds__(256) node_post_kernel(
    const float* __restrict__ agg_s, const float* __restrict__ agg_v,
    const float* __restrict__ l2ws, const float* __restrict__ l2wv,
    const float* __restrict__ sc_s, const float* __restrict__ sc_v,
    float* __restrict__ out) {
    const int n = blockIdx.x;
    __shared__ float sas[128];
    __shared__ float sav[384];
    __shared__ float sos[128];
    const int t = threadIdx.x;
    if (t < 128) sas[t] = agg_s[(size_t)n * 128 + t];
    sav[t] = agg_v[(size_t)n * 384 + t];
    if (t < 128) sav[256 + t] = agg_v[(size_t)n * 384 + 256 + t];
    __syncthreads();
    const float inv2 = 0.08838834764831845f;  // 1/sqrt(128)
    if (t < 128) {
        float acc = 0.0f;
        for (int u = 0; u < 128; ++u) acc += sas[u] * l2ws[u * 128 + t];
        sos[t] = acc * inv2 + sc_s[(size_t)n * 128 + t];
    }
    __syncthreads();
    if (t < 64) {
        out[(size_t)n * 256 + t] = silu_f(sos[t]);
    } else {
        int q = t - 64, v = q / 3, i = q % 3;
        float acc = 0.0f;
        for (int u = 0; u < 128; ++u) acc += sav[u * 3 + i] * l2wv[u * 64 + v];
        float ov = acc * inv2 + sc_v[(size_t)n * 192 + v * 3 + i];
        out[(size_t)n * 256 + 64 + q] = ov * silu_f(sos[64 + v]);
    }
}

// ---------------------------------------------------------------------------
extern "C" void kernel_launch(void* const* d_in, const int* in_sizes, int n_in,
                              void* d_out, int out_size, void* d_ws, size_t ws_size,
                              hipStream_t stream) {
    const float* node_features   = (const float*)d_in[0];
    const float* node_attributes = (const float*)d_in[1];
    const float* edge_sh         = (const float*)d_in[2];
    const float* edge_embedded   = (const float*)d_in[3];
    const float* lin1_ws         = (const float*)d_in[4];
    const float* lin1_wv         = (const float*)d_in[5];
    const float* mlp_w1          = (const float*)d_in[6];
    const float* mlp_w2          = (const float*)d_in[7];
    const float* mlp_w3          = (const float*)d_in[8];
    const float* lin2_ws         = (const float*)d_in[9];
    const float* lin2_wv         = (const float*)d_in[10];
    const float* sc_ws           = (const float*)d_in[11];
    const float* sc_wv           = (const float*)d_in[12];
    const int*   edge_src        = (const int*)d_in[13];
    const int*   edge_dst        = (const int*)d_in[14];

    float* ws    = (float*)d_ws;
    float* h0    = ws;                         // N*64
    float* h1    = h0 + (size_t)NN * 64;       // N*192
    float* sc_s  = h1 + (size_t)NN * 192;      // N*128
    float* sc_v  = sc_s + (size_t)NN * 128;    // N*192
    float* agg_s = sc_v + (size_t)NN * 192;    // N*128 (zeroed each call)
    float* agg_v = agg_s + (size_t)NN * 128;   // N*384 (zeroed each call)
    bf16_t* wf1  = (bf16_t*)(agg_v + (size_t)NN * 384);  // 4*512 bf16
    bf16_t* wf2  = wf1 + 4 * 512;                        // 8*512
    bf16_t* wf3  = wf2 + 8 * 512;                        // 32*512

    // zero the atomic accumulators (graph-capture-safe)
    hipMemsetAsync(agg_s, 0, (size_t)NN * (128 + 384) * sizeof(float), stream);

    // bf16 B-fragment pre-swizzle of MLP weights
    convert_weights_kernel<<<8,  256, 0, stream>>>(mlp_w1, 8,  1, 4,  wf1);
    convert_weights_kernel<<<16, 256, 0, stream>>>(mlp_w2, 64, 2, 4,  wf2);
    convert_weights_kernel<<<64, 256, 0, stream>>>(mlp_w3, 64, 2, 16, wf3);

    node_pre_kernel<<<NN, 256, 0, stream>>>(node_features, node_attributes,
                                            lin1_ws, lin1_wv, sc_ws, sc_wv,
                                            h0, h1, sc_s, sc_v);

    edge_kernel<<<EE / 128, 256, 0, stream>>>(edge_sh, edge_embedded,
                                              edge_src, edge_dst, h0, h1,
                                              wf1, wf2, wf3, agg_s, agg_v);

    node_post_kernel<<<NN, 256, 0, stream>>>(agg_s, agg_v, lin2_ws, lin2_wv,
                                             sc_s, sc_v, (float*)d_out);
}
